// mAtt_bci_12206297055698
// MI455X (gfx1250) — matrix operand order repaired and checked by reading, not compiled, not run on hardware
//
#include <hip/hip_runtime.h>

// MI455X / gfx1250 implementation.
//  - wave32 everywhere; 32-thread blocks for per-matrix (Jacobi / WMMA) work.
//  - v_wmma_f32_16x16x32_f16 used for the 20x20xK covariance Grams (f16 in, f32 acc).
//  - logm(rectify(expm(M))) algebraically collapsed to one eigh of M with
//    eigenvalue clamp at ln(1e-4) (same eigenvectors).
//  - deterministic fixed-order reductions (no float atomics).

typedef __attribute__((ext_vector_type(16))) _Float16 v16h;
typedef __attribute__((ext_vector_type(8)))  float    v8f;

namespace {
constexpr int BN   = 1024;   // batch
constexpr int C1   = 22;     // conv1 channels
constexpr int T1   = 438;    // time after conv1
constexpr int OC2  = 20;     // conv2 out channels == D_IN
constexpr int KW   = 12;     // conv2 kernel width
constexpr int TT2  = 439;    // time after conv2 (438 + 12 - 12 + 1)
constexpr int NEP  = 3;      // epochs
constexpr int DD   = 18;     // D_OUT
constexpr int NTRI = 171;    // triu(18) count
constexpr int FEAT = NTRI * NEP;  // 513
constexpr float BN_EPS   = 1e-5f;
constexpr float LOG_RECT = -9.210340371976184f;  // ln(1e-4)

// workspace layout (float offsets)
constexpr size_t WS_STATS = 0;                                    // 256 floats
constexpr size_t WS_H1  = 256;                                    // B*22*438
constexpr size_t WS_H2  = WS_H1  + (size_t)BN * C1 * T1;          // B*20*439
constexpr size_t WS_COV = WS_H2  + (size_t)BN * OC2 * TT2;        // B*3*20*20
constexpr size_t WS_QKV = WS_COV + (size_t)BN * NEP * OC2 * OC2;  // 3*B*3*18*18
constexpr size_t WS_LOG = WS_QKV + (size_t)3 * BN * NEP * DD * DD;
constexpr size_t WS_ML  = WS_LOG + (size_t)3 * BN * NEP * DD * DD;
constexpr size_t WS_TG  = WS_ML  + (size_t)BN * NEP * DD * DD;

constexpr int ST_B1SC = 44, ST_B1SH = 66, ST_B2SC = 128, ST_B2SH = 148;
}  // namespace

// ---------------------------------------------------------------- conv1 (22x22 GEMM per (b,t))
__global__ void k_conv1(const float* __restrict__ x, const float* __restrict__ w1,
                        const float* __restrict__ b1, float* __restrict__ h1) {
  __shared__ float xs[C1 * T1];
  __shared__ float wl[C1 * C1];
  const int b = blockIdx.x, tid = threadIdx.x;
  const size_t base = (size_t)b * (C1 * T1);
  for (int e = tid; e < C1 * T1; e += blockDim.x) xs[e] = x[base + e];
  for (int e = tid; e < C1 * C1; e += blockDim.x) wl[e] = w1[e];
  __syncthreads();
  for (int e = tid; e < C1 * T1; e += blockDim.x) {
    const int o = e / T1, t = e - o * T1;
    float acc = b1[o];
#pragma unroll
    for (int h = 0; h < C1; ++h) acc = fmaf(wl[o * C1 + h], xs[h * T1 + t], acc);
    h1[base + e] = acc;
  }
}

// ------------------------------------------- BN stats (one block per channel, deterministic tree)
__global__ void k_bnstats(const float* __restrict__ h, const float* __restrict__ g,
                          const float* __restrict__ bb, float* __restrict__ stats,
                          int nch, int tlen, int sc_off, int sh_off) {
  __shared__ float rs[256], rq[256];
  const int o = blockIdx.x, tid = threadIdx.x;
  const long n = (long)BN * tlen;
  float s = 0.f, q = 0.f;
  for (long idx = tid; idx < n; idx += blockDim.x) {
    const long b = idx / tlen;
    const int  t = (int)(idx - b * tlen);
    const float v = h[(size_t)b * nch * tlen + (size_t)o * tlen + t];
    s += v; q += v * v;
  }
  rs[tid] = s; rq[tid] = q;
  __syncthreads();
  for (int st = 128; st > 0; st >>= 1) {
    if (tid < st) { rs[tid] += rs[tid + st]; rq[tid] += rq[tid + st]; }
    __syncthreads();
  }
  if (tid == 0) {
    const float inv  = 1.0f / (float)n;
    const float mean = rs[0] * inv;
    const float var  = rq[0] * inv - mean * mean;
    const float sc   = g[o] * rsqrtf(var + BN_EPS);
    stats[sc_off + o] = sc;
    stats[sh_off + o] = bb[o] - mean * sc;
  }
}

// ---------------------------------------------------------------- conv2 (1x12, pad 6) + BN1 fused
__global__ void k_conv2(const float* __restrict__ h1, const float* __restrict__ w2,
                        const float* __restrict__ b2, const float* __restrict__ stats,
                        float* __restrict__ h2) {
  __shared__ float hs[C1 * T1];
  __shared__ float wl[OC2 * C1 * KW];
  __shared__ float scl[C1], shl[C1];
  const int b = blockIdx.x, tid = threadIdx.x;
  if (tid < C1) { scl[tid] = stats[ST_B1SC + tid]; shl[tid] = stats[ST_B1SH + tid]; }
  __syncthreads();
  const size_t base1 = (size_t)b * (C1 * T1);
  for (int e = tid; e < C1 * T1; e += blockDim.x) {
    const int i = e / T1;
    hs[e] = fmaf(h1[base1 + e], scl[i], shl[i]);
  }
  for (int e = tid; e < OC2 * C1 * KW; e += blockDim.x) wl[e] = w2[e];
  __syncthreads();
  const size_t base2 = (size_t)b * (OC2 * TT2);
  for (int e = tid; e < OC2 * TT2; e += blockDim.x) {
    const int o = e / TT2, t2 = e - o * TT2;
    float acc = b2[o];
    const int t0 = t2 - 6;
    for (int i = 0; i < C1; ++i) {
      const float* wr = &wl[(o * C1 + i) * KW];
      const float* hr = &hs[i * T1];
#pragma unroll
      for (int k = 0; k < KW; ++k) {
        const int p = t0 + k;
        if (p >= 0 && p < T1) acc = fmaf(wr[k], hr[p], acc);
      }
    }
    h2[base2 + e] = acc;
  }
}

// -------------------------- covariance Gram via WMMA: cov = G/tr(G) + 1e-5*I, G = S_c * S_c^T
// one wave32 per (b, epoch); 20x20 padded to 2x2 tiles of 16x16, K padded to 160 (5 steps of 32)
__global__ void k_cov(const float* __restrict__ h2, const float* __restrict__ stats,
                      float* __restrict__ cov) {
  __shared__ _Float16 segH[32][160];
  __shared__ float gram[OC2][OC2];
  __shared__ float sred;
  const int blk = blockIdx.x;
  const int b = blk / NEP, m = blk - b * NEP;
  const int lane = threadIdx.x;
  const int offs[3] = {0, 147, 293};
  const int lens[3] = {147, 146, 146};
  const int off = offs[m], L = lens[m];

  if (lane < OC2) {  // lane r owns row r: BN2 + center + f16 convert
    const int r = lane;
    const size_t base = (size_t)b * (OC2 * TT2) + (size_t)r * TT2 + off;
    const float sc = stats[ST_B2SC + r], sh = stats[ST_B2SH + r];
    float sum = 0.f;
    for (int k = 0; k < L; ++k) sum += fmaf(h2[base + k], sc, sh);
    const float mean = sum / (float)L;
    for (int k = 0; k < L; ++k)
      segH[r][k] = (_Float16)(fmaf(h2[base + k], sc, sh) - mean);
    for (int k = L; k < 160; ++k) segH[r][k] = (_Float16)0.f;
  } else {
    for (int k = 0; k < 160; ++k) segH[lane][k] = (_Float16)0.f;
  }
  __syncthreads();

  v8f c00 = {}, c01 = {}, c10 = {}, c11 = {};
  const int l  = lane & 15;
  const int hi = (lane >> 4) & 1;
  for (int ks = 0; ks < 5; ++ks) {
    const int k0 = ks * 32;
    v16h aLo, aHi, bLo, bHi;
    // A 16x32 f16 layout: lane<16: elems0-7=K0-7, elems8-15=K16-23; lane>=16: +8
#pragma unroll
    for (int j = 0; j < 8; ++j) {
      const int kA1 = k0 + hi * 8 + j;
      const int kA2 = kA1 + 16;
      aLo[j]     = segH[l][kA1];      aLo[j + 8] = segH[l][kA2];
      aHi[j]     = segH[l + 16][kA1]; aHi[j + 8] = segH[l + 16][kA2];
    }
    // B 32x16 f16 layout: col=l, same K order as A: elems0-7=K0-7, elems8-15=K16-23; lane>=16: +8. B[k][n]=S[n][k]
#pragma unroll
    for (int j = 0; j < 16; ++j) {
      const int kB = k0 + hi * 8 + (j & 7) + ((j >> 3) << 4);
      bLo[j] = segH[l][kB];
      bHi[j] = segH[l + 16][kB];
    }
    c00 = __builtin_amdgcn_wmma_f32_16x16x32_f16(false, aLo, false, bLo, (short)0, c00, false, false);
    c01 = __builtin_amdgcn_wmma_f32_16x16x32_f16(false, aLo, false, bHi, (short)0, c01, false, false);
    c10 = __builtin_amdgcn_wmma_f32_16x16x32_f16(false, aHi, false, bLo, (short)0, c10, false, false);
    c11 = __builtin_amdgcn_wmma_f32_16x16x32_f16(false, aHi, false, bHi, (short)0, c11, false, false);
  }
  // C/D layout: VGPR j -> (M=j+8*hi, N=l)
#pragma unroll
  for (int j = 0; j < 8; ++j) {
    const int row = j + hi * 8, col = l;
    if (row < OC2      && col < OC2)      gram[row][col]           = c00[j];
    if (row < OC2      && col + 16 < OC2) gram[row][col + 16]      = c01[j];
    if (row + 16 < OC2 && col < OC2)      gram[row + 16][col]      = c10[j];
    if (row + 16 < OC2 && col + 16 < OC2) gram[row + 16][col + 16] = c11[j];
  }
  __syncthreads();
  if (lane == 0) {
    float tr = 0.f;
    for (int i = 0; i < OC2; ++i) tr += gram[i][i];
    sred = 1.0f / tr;
  }
  __syncthreads();
  const float invtr = sred;
  float* outp = cov + (size_t)blk * (OC2 * OC2);
  for (int e = lane; e < OC2 * OC2; e += 32) {
    const int i = e / OC2, j = e - i * OC2;
    outp[e] = gram[i][j] * invtr + (i == j ? 1e-5f : 0.f);
  }
}

// ---------------------------------------------------------------- Q/K/V = W^T cov W  (f32)
__global__ void k_qkv(const float* __restrict__ cov, const float* __restrict__ Wq,
                      const float* __restrict__ Wk, const float* __restrict__ Wv,
                      float* __restrict__ qkv) {
  __shared__ float covL[OC2 * OC2], wL[OC2 * DD], tmpL[OC2 * DD];
  const int blk = blockIdx.x, tid = threadIdx.x;  // blk = b*3+m
  const float* src = cov + (size_t)blk * (OC2 * OC2);
  for (int e = tid; e < OC2 * OC2; e += blockDim.x) covL[e] = src[e];
  const float* Ws[3] = {Wq, Wk, Wv};
  for (int w = 0; w < 3; ++w) {
    __syncthreads();
    const float* W = Ws[w];
    for (int e = tid; e < OC2 * DD; e += blockDim.x) wL[e] = W[e];
    __syncthreads();
    for (int e = tid; e < OC2 * DD; e += blockDim.x) {  // tmp = cov @ W (20x18)
      const int i = e / DD, a = e - i * DD;
      float acc = 0.f;
      for (int j = 0; j < OC2; ++j) acc = fmaf(covL[i * OC2 + j], wL[j * DD + a], acc);
      tmpL[e] = acc;
    }
    __syncthreads();
    float* dst = qkv + ((size_t)w * BN * NEP + blk) * (DD * DD);
    for (int e = tid; e < DD * DD; e += blockDim.x) {  // out = W^T @ tmp (18x18)
      const int a = e / DD, c = e - a * DD;
      float acc = 0.f;
      for (int i = 0; i < OC2; ++i) acc = fmaf(wL[i * DD + a], tmpL[i * DD + c], acc);
      dst[e] = acc;
    }
  }
}

// ------------------- one-wave cyclic Jacobi eigh of 18x18 + spectral map reconstruction
// mode 0: f(s)=log(s)            (logm of SPD Q/K/V)
// mode 1: f(s)=max(s, ln(1e-4))  (collapsed expm->rectify->logm chain)
__global__ void k_eig(const float* __restrict__ src, float* __restrict__ dst, int mode) {
  __shared__ float Am[DD * 19], Vm[DD * 19], fs[DD];
  const int blk = blockIdx.x, lane = threadIdx.x;
  const float* in = src + (size_t)blk * (DD * DD);
  for (int e = lane; e < DD * DD; e += 32) {
    const int i = e / DD, j = e - i * DD;
    Am[i * 19 + j] = in[e];
  }
  for (int e = lane; e < DD * 19; e += 32) Vm[e] = 0.f;
  __syncthreads();
  if (lane < DD) Vm[lane * 19 + lane] = 1.f;
  __syncthreads();
#pragma unroll 1
  for (int sweep = 0; sweep < 8; ++sweep) {
#pragma unroll 1
    for (int p = 0; p < DD - 1; ++p) {
#pragma unroll 1
      for (int q = p + 1; q < DD; ++q) {
        const float app = Am[p * 19 + p], aqq = Am[q * 19 + q], apq = Am[p * 19 + q];
        float c = 1.f, s = 0.f;  // identity rotation if off-diagonal ~ 0 (branchless barriers)
        if (apq * apq > 1e-24f) {
          const float th = 0.5f * (aqq - app) / apq;
          const float t  = copysignf(1.0f, th) / (fabsf(th) + sqrtf(1.0f + th * th));
          c = 1.0f / sqrtf(1.0f + t * t);
          s = t * c;
        }
        if (lane < DD) {  // row update
          const float ap = Am[p * 19 + lane], aq = Am[q * 19 + lane];
          Am[p * 19 + lane] = c * ap - s * aq;
          Am[q * 19 + lane] = s * ap + c * aq;
        }
        __syncthreads();
        if (lane < DD) {  // column + eigenvector update
          const float ap = Am[lane * 19 + p], aq = Am[lane * 19 + q];
          Am[lane * 19 + p] = c * ap - s * aq;
          Am[lane * 19 + q] = s * ap + c * aq;
          const float vp = Vm[lane * 19 + p], vq = Vm[lane * 19 + q];
          Vm[lane * 19 + p] = c * vp - s * vq;
          Vm[lane * 19 + q] = s * vp + c * vq;
        }
        __syncthreads();
      }
    }
  }
  if (lane < DD) {
    const float sv = Am[lane * 19 + lane];
    fs[lane] = (mode == 0) ? logf(fmaxf(sv, 1e-30f)) : fmaxf(sv, LOG_RECT);
  }
  __syncthreads();
  float* out = dst + (size_t)blk * (DD * DD);
  for (int e = lane; e < DD * DD; e += 32) {
    const int i = e / DD, j = e - i * DD;
    float acc = 0.f;
#pragma unroll
    for (int k = 0; k < DD; ++k) acc = fmaf(Vm[i * 19 + k] * fs[k], Vm[j * 19 + k], acc);
    out[e] = acc;
  }
}

// ------------------- log-Euclidean attention: energy -> f -> softmax(axis=m) -> prob @ logV
__global__ void k_attn(const float* __restrict__ logqkv, float* __restrict__ ml) {
  __shared__ float fmat[9], prob[9];
  const int b = blockIdx.x, tid = threadIdx.x;
  const float* logQ = logqkv + ((size_t)0 * BN + b) * (NEP * DD * DD);
  const float* logK = logqkv + ((size_t)1 * BN + b) * (NEP * DD * DD);
  const float* logV = logqkv + ((size_t)2 * BN + b) * (NEP * DD * DD);
  if (tid < 9) {
    const int m = tid / 3, p = tid - m * 3;
    const float* km = logK + m * (DD * DD);
    const float* qp = logQ + p * (DD * DD);
    float e = 0.f;
    for (int r = 0; r < DD * DD; ++r) { const float d = km[r] - qp[r]; e = fmaf(d, d, e); }
    fmat[tid] = 1.0f / (1.0f + log1pf(e));
  }
  __syncthreads();
  if (tid < 3) {  // softmax over m for column p = tid
    const int p = tid;
    float mx = fmat[p];
    for (int m2 = 1; m2 < 3; ++m2) mx = fmaxf(mx, fmat[m2 * 3 + p]);
    float ex[3], ssum = 0.f;
    for (int m2 = 0; m2 < 3; ++m2) { ex[m2] = expf(fmat[m2 * 3 + p] - mx); ssum += ex[m2]; }
    for (int m2 = 0; m2 < 3; ++m2) prob[m2 * 3 + p] = ex[m2] / ssum;
  }
  __syncthreads();
  float* out = ml + (size_t)b * (NEP * DD * DD);
  for (int e = tid; e < NEP * DD * DD; e += blockDim.x) {
    const int p = e / (DD * DD), r = e - p * (DD * DD);
    float acc = 0.f;
    for (int i = 0; i < 3; ++i) acc = fmaf(prob[i * 3 + p], logV[i * (DD * DD) + r], acc);
    out[e] = acc;
  }
}

// ---------------------------------------------------------------- triu-gather + 513x4 head
__global__ void k_final(const float* __restrict__ tang, const float* __restrict__ lw,
                        const float* __restrict__ lb, float* __restrict__ out) {
  __shared__ float red[128 * 4];
  const int b = blockIdx.x, tid = threadIdx.x;
  float acc[4] = {0.f, 0.f, 0.f, 0.f};
  const float* tg = tang + (size_t)b * (NEP * DD * DD);
  for (int e = tid; e < FEAT; e += blockDim.x) {
    const int m = e / NTRI;
    int rem = e - m * NTRI, r = 0;
    while (rem >= DD - r) { rem -= DD - r; ++r; }
    const int c = r + rem;
    const float v = tg[m * (DD * DD) + r * DD + c];
#pragma unroll
    for (int j = 0; j < 4; ++j) acc[j] = fmaf(v, lw[j * FEAT + e], acc[j]);
  }
#pragma unroll
  for (int j = 0; j < 4; ++j) red[tid * 4 + j] = acc[j];
  __syncthreads();
  for (int st = 64; st > 0; st >>= 1) {
    if (tid < st) {
#pragma unroll
      for (int j = 0; j < 4; ++j) red[tid * 4 + j] += red[(tid + st) * 4 + j];
    }
    __syncthreads();
  }
  if (tid < 4) out[b * 4 + tid] = red[tid] + lb[tid];
}

// ---------------------------------------------------------------- host side
extern "C" void kernel_launch(void* const* d_in, const int* in_sizes, int n_in,
                              void* d_out, int out_size, void* d_ws, size_t ws_size,
                              hipStream_t stream) {
  (void)in_sizes; (void)n_in; (void)out_size; (void)ws_size;
  const float* x   = (const float*)d_in[0];
  const float* w1  = (const float*)d_in[1];
  const float* b1  = (const float*)d_in[2];
  const float* g1  = (const float*)d_in[3];
  const float* bb1 = (const float*)d_in[4];
  const float* w2  = (const float*)d_in[5];
  const float* b2  = (const float*)d_in[6];
  const float* g2  = (const float*)d_in[7];
  const float* bb2 = (const float*)d_in[8];
  const float* Wq  = (const float*)d_in[9];
  const float* Wk  = (const float*)d_in[10];
  const float* Wv  = (const float*)d_in[11];
  const float* lw  = (const float*)d_in[12];
  const float* lb  = (const float*)d_in[13];

  float* ws    = (float*)d_ws;
  float* stats = ws + WS_STATS;
  float* h1    = ws + WS_H1;
  float* h2    = ws + WS_H2;
  float* covb  = ws + WS_COV;
  float* qkvb  = ws + WS_QKV;
  float* logb  = ws + WS_LOG;
  float* mlb   = ws + WS_ML;
  float* tgb   = ws + WS_TG;

  k_conv1  <<<BN, 256, 0, stream>>>(x, w1, b1, h1);
  k_bnstats<<<C1, 256, 0, stream>>>(h1, g1, bb1, stats, C1, T1, ST_B1SC, ST_B1SH);
  k_conv2  <<<BN, 256, 0, stream>>>(h1, w2, b2, stats, h2);
  k_bnstats<<<OC2, 256, 0, stream>>>(h2, g2, bb2, stats, OC2, TT2, ST_B2SC, ST_B2SH);
  k_cov    <<<BN * NEP, 32, 0, stream>>>(h2, stats, covb);
  k_qkv    <<<BN * NEP, 128, 0, stream>>>(covb, Wq, Wk, Wv, qkvb);
  k_eig    <<<3 * BN * NEP, 32, 0, stream>>>(qkvb, logb, 0);
  k_attn   <<<BN, 128, 0, stream>>>(logb, mlb);
  k_eig    <<<BN * NEP, 32, 0, stream>>>(mlb, tgb, 1);
  k_final  <<<BN, 128, 0, stream>>>(tgb, lw, lb, (float*)d_out);
}
